// MyLLMFFNMoE_78718160601090
// MI455X (gfx1250) — matrix operand order repaired and checked by reading, not compiled, not run on hardware
//
#include <hip/hip_runtime.h>
#include <hip/hip_bf16.h>

typedef __bf16 bf16_t;
typedef __attribute__((ext_vector_type(16))) __bf16 v16bf;
typedef __attribute__((ext_vector_type(8)))  __bf16 v8bf;
typedef __attribute__((ext_vector_type(4)))  __bf16 v4bf;
typedef __attribute__((ext_vector_type(8)))  float  v8f;
typedef __attribute__((ext_vector_type(4)))  int    v4i;
typedef __attribute__((address_space(1))) v4i* gp128_t;   // global int4*
typedef __attribute__((address_space(3))) v4i* lp128_t;   // LDS int4*

#define H_DIM 2048
#define I_DIM 2048
#define E_R   7
#define NTOK  4096

#define TM 128
#define TN 128
#define TK 64
#define LDK 72   // padded LDS K-stride (elems): 144B rows, 16B aligned, conflict-free b128

#if defined(__HIP_DEVICE_COMPILE__) && defined(__gfx1250__) && \
    __has_builtin(__builtin_amdgcn_global_load_async_to_lds_b128) && \
    __has_builtin(__builtin_amdgcn_s_wait_asynccnt)
#define USE_ASYNC_LDS 1
#else
#define USE_ASYNC_LDS 0
#endif

// ---------------- router: logits -> top2 -> masked softmax -> probs[N, E_R] ----
__global__ __launch_bounds__(256) void router_kernel(
    const float* __restrict__ x, const float* __restrict__ Wr,
    const float* __restrict__ br, const float* __restrict__ gbias,
    float* __restrict__ probs)
{
  const int wave = threadIdx.x >> 5;
  const int lane = threadIdx.x & 31;
  const int n = blockIdx.x * 8 + wave;
  float acc[E_R];
  #pragma unroll
  for (int e = 0; e < E_R; ++e) acc[e] = 0.f;
  const float* xr = x + (size_t)n * H_DIM;
  for (int h = lane; h < H_DIM; h += 32) {
    float xv = xr[h];
    const float* wr = Wr + (size_t)h * E_R;
    #pragma unroll
    for (int e = 0; e < E_R; ++e) acc[e] += xv * wr[e];
  }
  #pragma unroll
  for (int e = 0; e < E_R; ++e) {
    float v = acc[e];
    #pragma unroll
    for (int off = 16; off > 0; off >>= 1) v += __shfl_down(v, off, 32);
    acc[e] = v;
  }
  if (lane == 0) {
    float logit[E_R], biased[E_R];
    #pragma unroll
    for (int e = 0; e < E_R; ++e) { logit[e] = acc[e] + br[e]; biased[e] = logit[e] + gbias[e]; }
    int i0 = 0;
    #pragma unroll
    for (int e = 1; e < E_R; ++e) if (biased[e] > biased[i0]) i0 = e;   // ties -> lowest idx
    int i1 = (i0 == 0) ? 1 : 0;
    #pragma unroll
    for (int e = 0; e < E_R; ++e) { if (e == i0 || e == i1) continue; if (biased[e] > biased[i1]) i1 = e; }
    float l0 = logit[i0], l1 = logit[i1];
    float m  = fmaxf(l0, l1);
    float e0 = __expf(l0 - m), e1 = __expf(l1 - m);
    float inv = 1.f / (e0 + e1);
    float* pr = probs + (size_t)n * E_R;
    #pragma unroll
    for (int e = 0; e < E_R; ++e) pr[e] = 0.f;
    pr[i0] = e0 * inv; pr[i1] = e1 * inv;
  }
}

// ---------------- fp32 -> bf16 vectorized conversion ---------------------------
__global__ __launch_bounds__(256) void cvt_kernel(const float4* __restrict__ s,
                                                  v4bf* __restrict__ d, int n4)
{
  int i = blockIdx.x * 256 + threadIdx.x;
  if (i < n4) {
    float4 f = s[i];
    v4bf o;
    o[0] = (bf16_t)f.x; o[1] = (bf16_t)f.y; o[2] = (bf16_t)f.z; o[3] = (bf16_t)f.w;
    d[i] = o;
  }
}

// ---------------- weight transform: fp32 [K][N] -> bf16 [N][K] (per z-matrix) --
__global__ __launch_bounds__(256) void wtrans_kernel(
    const float* __restrict__ src, bf16_t* __restrict__ dst, int K, int N)
{
  __shared__ bf16_t sT[64 * 68];   // 68-elem stride: 136B rows (8B aligned v4bf)
  const size_t moff = (size_t)K * N * blockIdx.z;
  const float* s = src + moff;
  bf16_t*      d = dst + moff;
  const int kb = blockIdx.x * 64, nb = blockIdx.y * 64;
  const int t  = threadIdx.x;
  const int r  = t >> 4;             // 0..15
  const int c4 = (t & 15) * 4;       // 0..60
  #pragma unroll
  for (int i = 0; i < 4; ++i) {
    int k = r + i * 16;
    float4 f = *(const float4*)(s + (size_t)(kb + k) * N + nb + c4);
    sT[(c4 + 0) * 68 + k] = (bf16_t)f.x;
    sT[(c4 + 1) * 68 + k] = (bf16_t)f.y;
    sT[(c4 + 2) * 68 + k] = (bf16_t)f.z;
    sT[(c4 + 3) * 68 + k] = (bf16_t)f.w;
  }
  __syncthreads();
  #pragma unroll
  for (int i = 0; i < 4; ++i) {
    int n = r + i * 16;
    v4bf v = *(const v4bf*)(&sT[n * 68 + c4]);
    *(v4bf*)(d + (size_t)(nb + n) * K + kb + c4) = v;   // coalesced along K
  }
}

// ---------------- tiled WMMA GEMM (bf16 x bf16 -> f32) with fused epilogues ----
// A  : [M][K] bf16 row-major (K contiguous)
// Bt : [N][K] bf16 row-major (K contiguous) == W^T
// MODE 0: out_bf16 = (A*B + bias)
// MODE 1: out_bf16 = silu(A*B + bias) * aux_bf16
// MODE 2: out_f32 (+)= p * (A*B + bias), p = probs[row, expert] or 1
template<int MODE>
__global__ __launch_bounds__(256) void moe_gemm(
    const bf16_t* __restrict__ A,
    const bf16_t* __restrict__ Bt,
    const float*  __restrict__ bias,
    const void*   __restrict__ aux,
    void* __restrict__ out,
    int M, int K, int N, int expert, int accumulate)
{
  __shared__ bf16_t sA[2][TM * LDK];
  __shared__ bf16_t sB[2][TN * LDK];

  const int tid  = threadIdx.x;
  const int lane = tid & 31;
  const int wave = tid >> 5;
  const int wm   = wave & 3;        // 32 rows per wave-M tile
  const int wn   = wave >> 2;       // 64 cols per wave-N tile
  const int m0   = blockIdx.y * TM;
  const int n0   = blockIdx.x * TN;

  v8f zero = {0.f,0.f,0.f,0.f,0.f,0.f,0.f,0.f};
  v8f acc[2][4];
  #pragma unroll
  for (int mi = 0; mi < 2; ++mi)
    #pragma unroll
    for (int ni = 0; ni < 4; ++ni) acc[mi][ni] = zero;

  const int kbA = (lane < 16) ? 0 : 16;   // A-frag K chunks, contiguous order k = 16h + s
  const int kbB = (lane < 16) ? 0 : 16;   // B-frag K chunks, same contiguous order as A
  const int l15 = lane & 31 & 15;
  const int nk  = K / TK;

  // copy geometry: 128 rows x 64 elems (bf16) = 1024 x 16B chunks, 4 per thread
  // q = tid + i*256 -> row = q>>3, kchunk = (q&7)*8
#if USE_ASYNC_LDS
  #define ISSUE_TILE(buf, kt)                                                      \
    {                                                                              \
      int kk0 = (kt) * TK;                                                         \
      _Pragma("unroll")                                                            \
      for (int i = 0; i < 4; ++i) {                                                \
        int q = tid + i * 256; int r = q >> 3; int kc = (q & 7) * 8;               \
        __builtin_amdgcn_global_load_async_to_lds_b128(                            \
            (gp128_t)(A + (size_t)(m0 + r) * K + kk0 + kc),                        \
            (lp128_t)(&sA[buf][r * LDK + kc]), 0, 0);                              \
        __builtin_amdgcn_global_load_async_to_lds_b128(                            \
            (gp128_t)(Bt + (size_t)(n0 + r) * K + kk0 + kc),                       \
            (lp128_t)(&sB[buf][r * LDK + kc]), 0, 0);                              \
      }                                                                            \
    }

  ISSUE_TILE(0, 0);
  for (int kt = 0; kt < nk; ++kt) {
    const int cur = kt & 1;
    __builtin_amdgcn_s_wait_asynccnt(0);   // this wave's tile-kt copies done
    __syncthreads();                       // all waves' copies done + prior reads done
    if (kt + 1 < nk) ISSUE_TILE(cur ^ 1, kt + 1);
#else
  v8bf rgA[4], rgB[4];
  #pragma unroll
  for (int i = 0; i < 4; ++i) {
    int q = tid + i * 256; int r = q >> 3; int kc = (q & 7) * 8;
    rgA[i] = *(const v8bf*)(A  + (size_t)(m0 + r) * K + kc);
    rgB[i] = *(const v8bf*)(Bt + (size_t)(n0 + r) * K + kc);
  }
  for (int kt = 0; kt < nk; ++kt) {
    const int cur = kt & 1;
    if (kt > 0) __syncthreads();
    #pragma unroll
    for (int i = 0; i < 4; ++i) {
      int q = tid + i * 256; int r = q >> 3; int kc = (q & 7) * 8;
      *(v8bf*)(&sA[cur][r * LDK + kc]) = rgA[i];
      *(v8bf*)(&sB[cur][r * LDK + kc]) = rgB[i];
    }
    __syncthreads();
    if (kt + 1 < nk) {
      int kk0 = (kt + 1) * TK;
      #pragma unroll
      for (int i = 0; i < 4; ++i) {
        int q = tid + i * 256; int r = q >> 3; int kc = (q & 7) * 8;
        rgA[i] = *(const v8bf*)(A  + (size_t)(m0 + r) * K + kk0 + kc);
        rgB[i] = *(const v8bf*)(Bt + (size_t)(n0 + r) * K + kk0 + kc);
      }
    }
#endif
    // ---- compute this k-tile: two K=32 sub-steps ----
    #pragma unroll
    for (int kk = 0; kk < 2; ++kk) {
      const int kb = kk * 32;
      v16bf afrag[2];
      #pragma unroll
      for (int mi = 0; mi < 2; ++mi) {
        int m = wm * 32 + mi * 16 + l15;
        union { v16bf v; v8bf h[2]; } u;
        u.h[0] = *(const v8bf*)(&sA[cur][m * LDK + kb + kbA]);
        u.h[1] = *(const v8bf*)(&sA[cur][m * LDK + kb + kbA + 8]);
        afrag[mi] = u.v;
      }
      v16bf bfrag[4];
      #pragma unroll
      for (int ni = 0; ni < 4; ++ni) {
        int nn = wn * 64 + ni * 16 + l15;
        union { v16bf v; v8bf h[2]; } u;
        u.h[0] = *(const v8bf*)(&sB[cur][nn * LDK + kb + kbB]);
        u.h[1] = *(const v8bf*)(&sB[cur][nn * LDK + kb + kbB + 8]);
        bfrag[ni] = u.v;
      }
      #pragma unroll
      for (int mi = 0; mi < 2; ++mi)
        #pragma unroll
        for (int ni = 0; ni < 4; ++ni)
          acc[mi][ni] = __builtin_amdgcn_wmma_f32_16x16x32_bf16(
              false, afrag[mi], false, bfrag[ni], (short)0, acc[mi][ni], false, false);
    }
  }

  // epilogue: D layout — VGPR r: lanes 0-15 -> M=r, lanes 16-31 -> M=8+r; N=lane&15
  const int row_base = m0 + wm * 32;
  const int col_base = n0 + wn * 64;
  const int rhi = (lane >= 16) ? 8 : 0;
  #pragma unroll
  for (int mi = 0; mi < 2; ++mi) {
    #pragma unroll
    for (int ni = 0; ni < 4; ++ni) {
      #pragma unroll
      for (int r = 0; r < 8; ++r) {
        int row = row_base + mi * 16 + rhi + r;
        int col = col_base + ni * 16 + l15;
        float v = acc[mi][ni][r] + bias[col];
        if (MODE == 0) {
          ((bf16_t*)out)[(size_t)row * N + col] = (bf16_t)v;
        } else if (MODE == 1) {
          const bf16_t* U = (const bf16_t*)aux;
          float uu = (float)U[(size_t)row * N + col];
          float s  = v / (1.f + __expf(-v));              // silu
          ((bf16_t*)out)[(size_t)row * N + col] = (bf16_t)(s * uu);
        } else {
          const float* probs = (const float*)aux;
          float p = probs ? probs[(size_t)row * E_R + expert] : 1.0f;
          float val = p * v;
          float* o = (float*)out + (size_t)row * N + col;
          if (accumulate) *o += val; else *o = val;
        }
      }
    }
  }
}

extern "C" void kernel_launch(void* const* d_in, const int* in_sizes, int n_in,
                              void* d_out, int out_size, void* d_ws, size_t ws_size,
                              hipStream_t stream) {
  const float* x     = (const float*)d_in[0];
  const float* Wr    = (const float*)d_in[1];
  const float* br    = (const float*)d_in[2];
  const float* gbias = (const float*)d_in[3];
  const float* Wup   = (const float*)d_in[4];
  const float* bup   = (const float*)d_in[5];
  const float* Wgate = (const float*)d_in[6];
  const float* bgate = (const float*)d_in[7];
  const float* Wdown = (const float*)d_in[8];
  const float* bdown = (const float*)d_in[9];
  const float* Wsu   = (const float*)d_in[10];
  const float* bsu   = (const float*)d_in[11];
  const float* Wsg   = (const float*)d_in[12];
  const float* bsg   = (const float*)d_in[13];
  const float* Wsd   = (const float*)d_in[14];
  const float* bsd   = (const float*)d_in[15];
  float* out = (float*)d_out;

  const size_t MAT = (size_t)2048 * 2048;   // elems per weight matrix
  char* ws = (char*)d_ws;
  float* probs = (float*)ws;
  size_t off = ((size_t)NTOK * E_R * sizeof(float) + 255) & ~(size_t)255;
  bf16_t* Xb = (bf16_t*)(ws + off); off += (size_t)NTOK * H_DIM * sizeof(bf16_t);
  bf16_t* Ub = (bf16_t*)(ws + off); off += (size_t)NTOK * I_DIM * sizeof(bf16_t);
  bf16_t* Gb = (bf16_t*)(ws + off); off += (size_t)NTOK * I_DIM * sizeof(bf16_t);
  bf16_t* WupT   = (bf16_t*)(ws + off); off += E_R * MAT * sizeof(bf16_t);
  bf16_t* WgateT = (bf16_t*)(ws + off); off += E_R * MAT * sizeof(bf16_t);
  bf16_t* WdownT = (bf16_t*)(ws + off); off += E_R * MAT * sizeof(bf16_t);
  bf16_t* WsuT   = (bf16_t*)(ws + off); off += MAT * sizeof(bf16_t);
  bf16_t* WsgT   = (bf16_t*)(ws + off); off += MAT * sizeof(bf16_t);
  bf16_t* WsdT   = (bf16_t*)(ws + off);

  router_kernel<<<NTOK / 8, 256, 0, stream>>>(x, Wr, br, gbias, probs);
  int n4 = NTOK * H_DIM / 4;
  cvt_kernel<<<(n4 + 255) / 256, 256, 0, stream>>>((const float4*)x, (v4bf*)Xb, n4);

  // transpose+convert all weights: fp32 [K][N] -> bf16 [N][K]
  dim3 tgridE(2048 / 64, 2048 / 64, E_R), tgrid1(2048 / 64, 2048 / 64, 1);
  wtrans_kernel<<<tgridE, 256, 0, stream>>>(Wup,   WupT,   H_DIM, I_DIM);
  wtrans_kernel<<<tgridE, 256, 0, stream>>>(Wgate, WgateT, I_DIM, I_DIM);
  wtrans_kernel<<<tgridE, 256, 0, stream>>>(Wdown, WdownT, I_DIM, H_DIM);
  wtrans_kernel<<<tgrid1, 256, 0, stream>>>(Wsu,   WsuT,   H_DIM, I_DIM);
  wtrans_kernel<<<tgrid1, 256, 0, stream>>>(Wsg,   WsgT,   I_DIM, I_DIM);
  wtrans_kernel<<<tgrid1, 256, 0, stream>>>(Wsd,   WsdT,   I_DIM, H_DIM);

  dim3 blk(256);
  dim3 grdI(I_DIM / TN, NTOK / TM);
  dim3 grdH(H_DIM / TN, NTOK / TM);

  for (int e = 0; e < E_R; ++e) {
    moe_gemm<0><<<grdI, blk, 0, stream>>>(Xb, WupT + e * MAT,
                                          bup + (size_t)e * I_DIM, nullptr, Ub,
                                          NTOK, H_DIM, I_DIM, e, 0);
    moe_gemm<1><<<grdI, blk, 0, stream>>>(Ub, WgateT + e * MAT,
                                          bgate + (size_t)e * I_DIM, Ub, Gb,
                                          NTOK, I_DIM, I_DIM, e, 0);
    moe_gemm<2><<<grdH, blk, 0, stream>>>(Gb, WdownT + e * MAT,
                                          bdown + (size_t)e * H_DIM, probs, out,
                                          NTOK, I_DIM, H_DIM, e, (e > 0) ? 1 : 0);
  }
  // shared expert (p = 1, accumulate into out)
  moe_gemm<0><<<grdI, blk, 0, stream>>>(Xb, WsuT, bsu, nullptr, Ub, NTOK, H_DIM, I_DIM, 0, 0);
  moe_gemm<1><<<grdI, blk, 0, stream>>>(Ub, WsgT, bsg, Ub, Gb, NTOK, I_DIM, I_DIM, 0, 0);
  moe_gemm<2><<<grdH, blk, 0, stream>>>(Gb, WsdT, bsd, nullptr, out, NTOK, I_DIM, H_DIM, 0, 1);
}
